// SupervisedRefiner_36936718745723
// MI455X (gfx1250) — matrix operand order repaired and checked by reading, not compiled, not run on hardware
//
#include <hip/hip_runtime.h>
#include <hip/hip_bf16.h>
#include <math.h>
#include <stdint.h>

#define HID 1024
#define HEADS 16
#define HEAD_DIM 64
#define SEQ 1024
#define BATCH 2

typedef __attribute__((ext_vector_type(16))) __bf16 v16bf;
typedef __attribute__((ext_vector_type(8)))  __bf16 v8bf;
typedef __attribute__((ext_vector_type(8)))  float  v8f;

// ---------------------------------------------------------------------------
// Fragment loaders (CDNA5 WMMA 16-bit layouts, wave32)
// A 16x32: lane&15 = M row; elems 0..7 at koff=(lane>>4)*8, elems 8..15 at koff+16
// B 32x16: lane&15 = N col; elems 0..7 at koff=(lane>>4)*8, elems 8..15 at koff+16
// ---------------------------------------------------------------------------
static __device__ __forceinline__ v16bf load_a_frag(const __bf16* __restrict__ base,
                                                    int ld, int row, int k, int lane) {
  const __bf16* p = base + (size_t)(row + (lane & 15)) * ld + k + ((lane >> 4) << 3);
  v8bf lo = *(const v8bf*)(p);
  v8bf hi = *(const v8bf*)(p + 16);
  v16bf f;
#pragma unroll
  for (int i = 0; i < 8; ++i) { f[i] = lo[i]; f[i + 8] = hi[i]; }
  return f;
}

static __device__ __forceinline__ v16bf load_b_frag(const __bf16* __restrict__ base,
                                                    int ld, int col, int k, int lane) {
  const __bf16* p = base + (size_t)(col + (lane & 15)) * ld + k + ((lane >> 4) << 3);
  v8bf lo = *(const v8bf*)(p);
  v8bf hi = *(const v8bf*)(p + 16);
  v16bf f;
#pragma unroll
  for (int i = 0; i < 8; ++i) { f[i] = lo[i]; f[i + 8] = hi[i]; }
  return f;
}

// read four A fragments (64 rows) from an LDS-staged 64x32 bf16 tile
static __device__ __forceinline__ void lds_a_frags(const __bf16* lb, int lane,
                                                   v16bf (&afr)[4]) {
#pragma unroll
  for (int i = 0; i < 4; ++i) {
    const __bf16* p = lb + (16 * i + (lane & 15)) * 32 + ((lane >> 4) << 3);
    v8bf lo = *(const v8bf*)p;
    v8bf hi = *(const v8bf*)(p + 16);
#pragma unroll
    for (int e = 0; e < 8; ++e) { afr[i][e] = lo[e]; afr[i][e + 8] = hi[e]; }
  }
}

static __device__ __forceinline__ void wmma_4x4(v8f (&acc)[4][4], const v16bf (&afr)[4],
                                                const v16bf (&bfr)[4]) {
#pragma unroll
  for (int i = 0; i < 4; ++i)
#pragma unroll
    for (int j = 0; j < 4; ++j)
      acc[i][j] = __builtin_amdgcn_wmma_f32_16x16x32_bf16(
          false, afr[i], false, bfr[j], (short)0, acc[i][j], false, false);
}

// ---------------------------------------------------------------------------
// fp32 -> bf16 convert
// ---------------------------------------------------------------------------
__global__ void cvt_f32_to_bf16(const float* __restrict__ in, __bf16* __restrict__ out, int n) {
  int i = blockIdx.x * blockDim.x + threadIdx.x;
  if (i < n) out[i] = (__bf16)in[i];
}

// concat three 1024-float bias vectors into one 3072 vector
__global__ void concat3_bias(const float* __restrict__ a, const float* __restrict__ b,
                             const float* __restrict__ c, float* __restrict__ o) {
  int i = blockIdx.x * blockDim.x + threadIdx.x;  // 3072
  float v = (i < HID) ? a[i] : ((i < 2 * HID) ? b[i - HID] : c[i - 2 * HID]);
  o[i] = v;
}

// ---------------------------------------------------------------------------
// Row LayerNorm (D=1024), fp32 in -> bf16 out. One block (256 thr) per row.
// ---------------------------------------------------------------------------
__global__ __launch_bounds__(256)
void layernorm_to_bf16(const float* __restrict__ x, const float* __restrict__ g,
                       const float* __restrict__ b, __bf16* __restrict__ out) {
  __shared__ float red[8];
  const int row = blockIdx.x;
  const float* xr = x + (size_t)row * HID;
  float vals[4];
  float s = 0.f;
#pragma unroll
  for (int i = 0; i < 4; ++i) { vals[i] = xr[threadIdx.x + i * 256]; s += vals[i]; }
#pragma unroll
  for (int o = 16; o; o >>= 1) s += __shfl_xor(s, o, 32);
  if ((threadIdx.x & 31) == 0) red[threadIdx.x >> 5] = s;
  __syncthreads();
  float tot = 0.f;
#pragma unroll
  for (int w = 0; w < 8; ++w) tot += red[w];
  const float mean = tot * (1.0f / HID);
  __syncthreads();
  float vs = 0.f;
#pragma unroll
  for (int i = 0; i < 4; ++i) { float d = vals[i] - mean; vs += d * d; }
#pragma unroll
  for (int o = 16; o; o >>= 1) vs += __shfl_xor(vs, o, 32);
  if ((threadIdx.x & 31) == 0) red[threadIdx.x >> 5] = vs;
  __syncthreads();
  float vtot = 0.f;
#pragma unroll
  for (int w = 0; w < 8; ++w) vtot += red[w];
  const float rstd = rsqrtf(vtot * (1.0f / HID) + 1e-5f);
#pragma unroll
  for (int i = 0; i < 4; ++i) {
    const int c = threadIdx.x + i * 256;
    out[(size_t)row * HID + c] = (__bf16)((vals[i] - mean) * rstd * g[c] + b[c]);
  }
}

// ---------------------------------------------------------------------------
// WMMA GEMM: C[M,N] = A[M,K](bf16,row) x W[N,K](bf16,row)^T + bias (+gelu)(+resid)
// Block = 128 thr = 4 waves (2x2), block tile 128x128, wave tile 64x64.
// A tile staged via async global->LDS copies (double buffered); B fragments
// double buffered in registers. k-loop manually unrolled x2 so all buffer
// indices are compile-time constants (dynamic indexing spills to scratch).
// ---------------------------------------------------------------------------
template<int GELU, int RESID>
__global__ __launch_bounds__(128, 1)
void gemm_bf16_nt(const __bf16* __restrict__ A, const __bf16* __restrict__ W,
                  const float* __restrict__ bias, const float* __restrict__ resid,
                  float* __restrict__ outf, __bf16* __restrict__ outb,
                  int M, int N, int K) {
  // 4 waves x 2 buffers x (64 rows x 32 cols bf16) = 32 KB
  __shared__ __align__(16) __bf16 asmem[4][2][64 * 32];
  const int lane = threadIdx.x & 31;
  const int wave = threadIdx.x >> 5;
  const int row0 = blockIdx.y * 128 + (wave >> 1) * 64;
  const int col0 = blockIdx.x * 128 + (wave & 1) * 64;

  // async fill pattern: issue t (0..7) covers rows t*8 + (lane>>2), 16B chunk (lane&3)
  const int arow = lane >> 2;
  const int acol = (lane & 3) * 8;  // bf16 elems
  const __bf16* agbase = A + (size_t)(row0 + arow) * K + acol;
  const unsigned albase =
      (unsigned)(uintptr_t)(&asmem[wave][0][0]) + (unsigned)((arow * 32 + acol) * 2);

#define GEMM_FILL(bufidx, kofs)                                                        \
  do {                                                                                 \
    const __bf16* _g = agbase + (kofs);                                                \
    const unsigned _l = albase + (unsigned)(bufidx) * (64 * 32 * 2);                   \
    _Pragma("unroll")                                                                  \
    for (int _t = 0; _t < 8; ++_t) {                                                   \
      asm volatile("global_load_async_to_lds_b128 %0, %1, off"                         \
                   :: "v"(_l + _t * (8 * 32 * 2)), "v"(_g + (size_t)_t * 8 * K)        \
                   : "memory");                                                        \
    }                                                                                  \
  } while (0)

  v8f acc[4][4] = {};
  v16bf bfrA[4], bfrB[4];
#pragma unroll
  for (int j = 0; j < 4; ++j) bfrA[j] = load_b_frag(W, K, col0 + 16 * j, 0, lane);
  GEMM_FILL(0, 0);
  GEMM_FILL(1, 32);

  const __bf16* lb0 = &asmem[wave][0][0];
  const __bf16* lb1 = &asmem[wave][1][0];

  for (int k = 0; k < K; k += 64) {  // K is a multiple of 64
    // ---- half 1: consume LDS buf0 + bfrA (k); prefetch bfrB (k+32) ----
    {
#pragma unroll
      for (int j = 0; j < 4; ++j) bfrB[j] = load_b_frag(W, K, col0 + 16 * j, k + 32, lane);
      asm volatile("s_wait_asynccnt 0x8" ::: "memory");  // buf1(k+32) may stay in flight
      v16bf afr[4];
      lds_a_frags(lb0, lane, afr);
      wmma_4x4(acc, afr, bfrA);
      if (k + 64 < K) {
        asm volatile("s_wait_dscnt 0x0" ::: "memory");
        GEMM_FILL(0, k + 64);
      }
    }
    // ---- half 2: consume LDS buf1 + bfrB (k+32); prefetch bfrA (k+64) ----
    {
      if (k + 64 < K) {
#pragma unroll
        for (int j = 0; j < 4; ++j) bfrA[j] = load_b_frag(W, K, col0 + 16 * j, k + 64, lane);
        asm volatile("s_wait_asynccnt 0x8" ::: "memory");  // buf0(k+64) in flight
      } else {
        asm volatile("s_wait_asynccnt 0x0" ::: "memory");
      }
      v16bf afr[4];
      lds_a_frags(lb1, lane, afr);
      wmma_4x4(acc, afr, bfrB);
      if (k + 96 < K) {
        asm volatile("s_wait_dscnt 0x0" ::: "memory");
        GEMM_FILL(1, k + 96);
      }
    }
  }
#undef GEMM_FILL

  const int mo = ((lane >> 4) << 3);
  const int nc = lane & 15;
#pragma unroll
  for (int i = 0; i < 4; ++i)
#pragma unroll
    for (int j = 0; j < 4; ++j) {
      const int n = col0 + 16 * j + nc;
      const float bn = bias[n];
#pragma unroll
      for (int r = 0; r < 8; ++r) {
        const int m = row0 + 16 * i + mo + r;
        float v = acc[i][j][r] + bn;
        if (GELU) v = 0.5f * v * (1.0f + erff(v * 0.70710678118654752f));
        if (RESID) v += resid[(size_t)m * N + n];
        if (outf) outf[(size_t)m * N + n] = v;
        if (outb) outb[(size_t)m * N + n] = (__bf16)v;
      }
    }
}

// ---------------------------------------------------------------------------
// V transpose out of packed QKV: [B*N, 3*HID](col 2048..3071) -> [B,H,Dh,N] bf16
// ---------------------------------------------------------------------------
__global__ void transpose_v_kernel(const __bf16* __restrict__ qkv, __bf16* __restrict__ vT) {
  const int i = blockIdx.x * blockDim.x + threadIdx.x;  // B*H*Dh*N = 2^21
  const int n = i & (SEQ - 1);
  const int d = (i >> 10) & (HEAD_DIM - 1);
  const int h = (i >> 16) & (HEADS - 1);
  const int b = i >> 20;
  vT[i] = qkv[((size_t)(b * SEQ + n)) * (3 * HID) + 2 * HID + h * HEAD_DIM + d];
}

// ---------------------------------------------------------------------------
// Fused attention (flash-style, one wave = 16 query rows, full key sweep).
// K fragments double buffered across key chunks with a manually 2x-unrolled
// loop (constant buffer names -> no scratch). Post-softmax scale folded into
// the logit as +ln(scale) (scale >= 0.25 > 0 always).
// ---------------------------------------------------------------------------
__global__ __launch_bounds__(256, 1)
void attention_fused(const __bf16* __restrict__ qkv, const __bf16* __restrict__ vT,
                     const float* __restrict__ adjacency,
                     const float* __restrict__ positions,
                     const float* __restrict__ edge,
                     __bf16* __restrict__ att) {
  __shared__ __align__(16) __bf16 pt[8][16 * 32];
  const int lane = threadIdx.x & 31;
  const int wave = threadIdx.x >> 5;
  const int tile = blockIdx.x & 7;
  const int h = (blockIdx.x >> 3) & 15;
  const int b = blockIdx.x >> 7;
  const int hg = h >> 2;

  const int ldq = 3 * HID;
  const int qrow0 = tile * 128 + wave * 16;
  const __bf16* qbase = qkv + (size_t)b * SEQ * ldq + h * HEAD_DIM;
  const __bf16* kbase = qkv + (size_t)b * SEQ * ldq + HID + h * HEAD_DIM;
  const __bf16* vbase = vT + ((size_t)(b * HEADS + h) * HEAD_DIM) * SEQ;

  const v16bf qa0 = load_a_frag(qbase, ldq, qrow0, 0, lane);
  const v16bf qa1 = load_a_frag(qbase, ldq, qrow0, 32, lane);

  const int mo = ((lane >> 4) << 3);
  const int nc = lane & 15;

  float rmax[8], rsum[8];
#pragma unroll
  for (int r = 0; r < 8; ++r) { rmax[r] = -1e30f; rsum[r] = 0.f; }
  v8f oacc[4] = {};

  float qpx[8], qpy[8];
  if (hg == 0) {
#pragma unroll
    for (int r = 0; r < 8; ++r) {
      const int qr = qrow0 + mo + r;
      qpx[r] = positions[((size_t)b * SEQ + qr) * 2 + 0];
      qpy[r] = positions[((size_t)b * SEQ + qr) * 2 + 1];
    }
  }

  const float inv_sqrt_d = 0.125f;
  const float inv2s2 = 1.0f / (2.0f * 0.22f * 0.22f);
  __bf16* pw = pt[wave];

  // process one 32-key chunk with the given (pre-loaded) K fragments
  auto proc = [&](int j0c, const v16bf (&kfc)[4]) {
    v8f s0 = {}, s1 = {};
    s0 = __builtin_amdgcn_wmma_f32_16x16x32_bf16(false, qa0, false, kfc[0], (short)0, s0, false, false);
    s0 = __builtin_amdgcn_wmma_f32_16x16x32_bf16(false, qa1, false, kfc[1], (short)0, s0, false, false);
    s1 = __builtin_amdgcn_wmma_f32_16x16x32_bf16(false, qa0, false, kfc[2], (short)0, s1, false, false);
    s1 = __builtin_amdgcn_wmma_f32_16x16x32_bf16(false, qa1, false, kfc[3], (short)0, s1, false, false);

    const int cL = j0c + nc;
    const int cH = cL + 16;
    float yL[8], yH[8];  // logit + bias + ln(scale)

    if (hg == 0) {
      const float kxL = positions[((size_t)b * SEQ + cL) * 2 + 0];
      const float kyL = positions[((size_t)b * SEQ + cL) * 2 + 1];
      const float kxH = positions[((size_t)b * SEQ + cH) * 2 + 0];
      const float kyH = positions[((size_t)b * SEQ + cH) * 2 + 1];
#pragma unroll
      for (int r = 0; r < 8; ++r) {
        const float dxl = qpx[r] - kxL, dyl = qpy[r] - kyL;
        const float cl = __expf(-(dxl * dxl + dyl * dyl) * inv2s2);
        const float dxh = qpx[r] - kxH, dyh = qpy[r] - kyH;
        const float ch = __expf(-(dxh * dxh + dyh * dyh) * inv2s2);
        yL[r] = s0[r] * inv_sqrt_d + (2.f * cl - 1.f) + __logf(0.25f + 0.75f * cl);
        yH[r] = s1[r] * inv_sqrt_d + (2.f * ch - 1.f) + __logf(0.25f + 0.75f * ch);
      }
    } else if (hg == 1) {
      const float* adjb = adjacency + (size_t)b * SEQ * SEQ;
#pragma unroll
      for (int r = 0; r < 8; ++r) {
        const int qr = qrow0 + mo + r;
        const float al = adjb[(size_t)qr * SEQ + cL];
        const float ah = adjb[(size_t)qr * SEQ + cH];
        yL[r] = s0[r] * inv_sqrt_d + (2.f * al - 1.f) + __logf(0.25f + 0.75f * al);
        yH[r] = s1[r] * inv_sqrt_d + (2.f * ah - 1.f) + __logf(0.25f + 0.75f * ah);
      }
    } else if (hg == 2) {
      const float* eb = edge + (((size_t)b * 4 + (h & 3)) * SEQ) * SEQ;
#pragma unroll
      for (int r = 0; r < 8; ++r) {
        const int qr = qrow0 + mo + r;
        yL[r] = s0[r] * inv_sqrt_d + eb[(size_t)qr * SEQ + cL];
        yH[r] = s1[r] * inv_sqrt_d + eb[(size_t)qr * SEQ + cH];
      }
    } else {
#pragma unroll
      for (int r = 0; r < 8; ++r) {
        yL[r] = s0[r] * inv_sqrt_d;
        yH[r] = s1[r] * inv_sqrt_d;
      }
    }

    // plain flash online softmax over the 32-key chunk
#pragma unroll
    for (int r = 0; r < 8; ++r) {
      float cm = fmaxf(yL[r], yH[r]);
#pragma unroll
      for (int o = 8; o; o >>= 1) cm = fmaxf(cm, __shfl_xor(cm, o, 32));
      const float nm = fmaxf(rmax[r], cm);
      const float corr = __expf(rmax[r] - nm);
      rmax[r] = nm;
      const float pL = __expf(yL[r] - nm);
      const float pH = __expf(yH[r] - nm);
      float ps = pL + pH;
#pragma unroll
      for (int o = 8; o; o >>= 1) ps += __shfl_xor(ps, o, 32);
      rsum[r] = rsum[r] * corr + ps;
#pragma unroll
      for (int j = 0; j < 4; ++j) oacc[j][r] *= corr;
      pw[(mo + r) * 32 + nc] = (__bf16)pL;
      pw[(mo + r) * 32 + nc + 16] = (__bf16)pH;
    }

    // V fragments for this chunk: issued here so they fly under the LDS wait
    v16bf vf[4];
#pragma unroll
    for (int j = 0; j < 4; ++j) vf[j] = load_b_frag(vbase, SEQ, j * 16, j0c, lane);

    __builtin_amdgcn_wave_barrier();
    asm volatile("s_wait_dscnt 0x0" ::: "memory");
    __builtin_amdgcn_wave_barrier();

    v16bf pa;
    {
      const __bf16* pp = pw + (size_t)nc * 32 + ((lane >> 4) << 3);
      v8bf lo = *(const v8bf*)pp;
      v8bf hi = *(const v8bf*)(pp + 16);
#pragma unroll
      for (int i = 0; i < 8; ++i) { pa[i] = lo[i]; pa[i + 8] = hi[i]; }
    }

#pragma unroll
    for (int j = 0; j < 4; ++j)
      oacc[j] = __builtin_amdgcn_wmma_f32_16x16x32_bf16(
          false, pa, false, vf[j], (short)0, oacc[j], false, false);
    __builtin_amdgcn_wave_barrier();
  };

  // K fragments: two statically named banks, loop unrolled x2 (SEQ % 64 == 0)
  v16bf kfA[4], kfB[4];
  kfA[0] = load_b_frag(kbase, ldq, 0, 0, lane);
  kfA[1] = load_b_frag(kbase, ldq, 0, 32, lane);
  kfA[2] = load_b_frag(kbase, ldq, 16, 0, lane);
  kfA[3] = load_b_frag(kbase, ldq, 16, 32, lane);

  for (int j0 = 0; j0 < SEQ; j0 += 64) {
    // prefetch chunk j0+32 into bank B (always in range: j0 <= SEQ-64)
    kfB[0] = load_b_frag(kbase, ldq, j0 + 32, 0, lane);
    kfB[1] = load_b_frag(kbase, ldq, j0 + 32, 32, lane);
    kfB[2] = load_b_frag(kbase, ldq, j0 + 48, 0, lane);
    kfB[3] = load_b_frag(kbase, ldq, j0 + 48, 32, lane);
    proc(j0, kfA);

    if (j0 + 64 < SEQ) {
      kfA[0] = load_b_frag(kbase, ldq, j0 + 64, 0, lane);
      kfA[1] = load_b_frag(kbase, ldq, j0 + 64, 32, lane);
      kfA[2] = load_b_frag(kbase, ldq, j0 + 80, 0, lane);
      kfA[3] = load_b_frag(kbase, ldq, j0 + 80, 32, lane);
    }
    proc(j0 + 32, kfB);
  }

#pragma unroll
  for (int r = 0; r < 8; ++r) {
    const float inv = 1.0f / fmaxf(rsum[r], 1e-6f);
    const int qr = qrow0 + mo + r;
#pragma unroll
    for (int j = 0; j < 4; ++j) {
      const int d = j * 16 + nc;
      att[((size_t)b * SEQ + qr) * HID + h * HEAD_DIM + d] = (__bf16)(oacc[j][r] * inv);
    }
  }
}

// ---------------------------------------------------------------------------
// Launcher
// ---------------------------------------------------------------------------
extern "C" void kernel_launch(void* const* d_in, const int* in_sizes, int n_in,
                              void* d_out, int out_size, void* d_ws, size_t ws_size,
                              hipStream_t stream) {
  (void)in_sizes; (void)n_in; (void)out_size; (void)ws_size;

  const float* hidden    = (const float*)d_in[0];
  const float* adjacency = (const float*)d_in[1];
  const float* positions = (const float*)d_in[2];
  const float* edge      = (const float*)d_in[3];
  const float* Wq = (const float*)d_in[4];   const float* bq = (const float*)d_in[5];
  const float* Wk = (const float*)d_in[6];   const float* bk = (const float*)d_in[7];
  const float* Wv = (const float*)d_in[8];   const float* bv = (const float*)d_in[9];
  const float* Wo = (const float*)d_in[10];  const float* bo = (const float*)d_in[11];
  const float* ln1_g = (const float*)d_in[12]; const float* ln1_b = (const float*)d_in[13];
  const float* ln2_g = (const float*)d_in[14]; const float* ln2_b = (const float*)d_in[15];
  const float* Wff1 = (const float*)d_in[16]; const float* bff1 = (const float*)d_in[17];
  const float* Wff2 = (const float*)d_in[18]; const float* bff2 = (const float*)d_in[19];
  float* out = (float*)d_out;

  char* ws = (char*)d_ws;
  size_t off = 0;
  auto alloc = [&](size_t bytes) -> void* {
    void* p = ws + off;
    off = (off + bytes + 255) & ~(size_t)255;
    return p;
  };

  const size_t T = (size_t)BATCH * SEQ;  // 2048 token rows

  __bf16* wqkv_b = (__bf16*)alloc((size_t)3 * HID * HID * 2);  // [3072,1024]
  __bf16* wo_b   = (__bf16*)alloc((size_t)HID * HID * 2);
  __bf16* wff1_b = (__bf16*)alloc((size_t)4 * HID * HID * 2);
  __bf16* wff2_b = (__bf16*)alloc((size_t)4 * HID * HID * 2);
  float*  bqkv   = (float*)alloc((size_t)3 * HID * 4);
  __bf16* normA  = (__bf16*)alloc(T * HID * 2);
  __bf16* qkvbuf = (__bf16*)alloc(T * 3 * HID * 2);            // [2048,3072]
  __bf16* vTbuf  = (__bf16*)alloc(T * HID * 2);
  __bf16* attbuf = (__bf16*)alloc(T * HID * 2);
  float*  hres   = (float*)alloc(T * HID * 4);
  __bf16* n2buf  = (__bf16*)alloc(T * HID * 2);
  __bf16* f1buf  = (__bf16*)alloc(T * 4 * HID * 2);

  const int TPB = 256;
  cvt_f32_to_bf16<<<(HID * HID) / TPB, TPB, 0, stream>>>(Wq, wqkv_b, HID * HID);
  cvt_f32_to_bf16<<<(HID * HID) / TPB, TPB, 0, stream>>>(Wk, wqkv_b + (size_t)HID * HID, HID * HID);
  cvt_f32_to_bf16<<<(HID * HID) / TPB, TPB, 0, stream>>>(Wv, wqkv_b + (size_t)2 * HID * HID, HID * HID);
  cvt_f32_to_bf16<<<(HID * HID) / TPB, TPB, 0, stream>>>(Wo, wo_b, HID * HID);
  cvt_f32_to_bf16<<<(4 * HID * HID) / TPB, TPB, 0, stream>>>(Wff1, wff1_b, 4 * HID * HID);
  cvt_f32_to_bf16<<<(4 * HID * HID) / TPB, TPB, 0, stream>>>(Wff2, wff2_b, 4 * HID * HID);
  concat3_bias<<<(3 * HID) / TPB, TPB, 0, stream>>>(bq, bk, bv, bqkv);

  layernorm_to_bf16<<<(int)T, 256, 0, stream>>>(hidden, ln1_g, ln1_b, normA);

  // fused QKV projection: [2048,1024] x [3072,1024]^T
  dim3 gqkv(3 * HID / 128, (int)T / 128);  // (24, 16)
  gemm_bf16_nt<0, 0><<<gqkv, 128, 0, stream>>>(normA, wqkv_b, bqkv, nullptr, nullptr, qkvbuf,
                                               (int)T, 3 * HID, HID);

  transpose_v_kernel<<<(int)(T * HID) / TPB, TPB, 0, stream>>>(qkvbuf, vTbuf);

  attention_fused<<<BATCH * HEADS * (SEQ / 128), 256, 0, stream>>>(
      qkvbuf, vTbuf, adjacency, positions, edge, attbuf);

  dim3 g1(HID / 128, (int)T / 128);  // (8, 16)
  gemm_bf16_nt<0, 1><<<g1, 128, 0, stream>>>(attbuf, wo_b, bo, hidden, hres, nullptr,
                                             (int)T, HID, HID);

  layernorm_to_bf16<<<(int)T, 256, 0, stream>>>(hres, ln2_g, ln2_b, n2buf);

  dim3 g2(4 * HID / 128, (int)T / 128);  // (32, 16)
  gemm_bf16_nt<1, 0><<<g2, 128, 0, stream>>>(n2buf, wff1_b, bff1, nullptr, nullptr, f1buf,
                                             (int)T, 4 * HID, HID);

  gemm_bf16_nt<0, 1><<<g1, 128, 0, stream>>>(f1buf, wff2_b, bff2, hres, out, nullptr,
                                             (int)T, HID, 4 * HID);
}
